// EdgeResBlock_88605175316742
// MI455X (gfx1250) — matrix operand order repaired and checked by reading, not compiled, not run on hardware
//
#include <hip/hip_runtime.h>

// ---------------------------------------------------------------------------
// EdgeResBlock on MI455X (gfx1250): 4x implicit-GEMM 3x3 convs (2 PAC, 2 plain)
// via v_wmma_f32_16x16x32_bf16.  Roofline: ~39 GF of matmul vs ~200MB of L2-
// resident traffic (192MB L2, 23.3 TB/s HBM) -> compute-bound on the WMMA path.
//
// v3: fix async-to-LDS builtin arg types: it takes int4-vector pointers
// (addrspace(1) source, addrspace(3) LDS destination), not void*.
// ---------------------------------------------------------------------------

#ifndef __has_builtin
#define __has_builtin(x) 0
#endif
#if __has_builtin(__builtin_amdgcn_global_load_async_to_lds_b128)
#define USE_ASYNC_LDS 1
#else
#define USE_ASYNC_LDS 0
#endif

typedef __attribute__((ext_vector_type(16))) __bf16 v16bf;
typedef __attribute__((ext_vector_type(8)))  float  v8f;
typedef int v4i __attribute__((vector_size(16)));

union BF16x16 { v16bf v; uint4 q[2]; unsigned short u[16]; };

static __device__ __forceinline__ float bf2f(unsigned short u) {
  union { unsigned int i; float f; } c; c.i = ((unsigned int)u) << 16; return c.f;
}
static __device__ __forceinline__ unsigned short f2bf(float f) {
  union { float f; unsigned int i; } c; c.f = f;
  unsigned int r = c.i + 0x7fffu + ((c.i >> 16) & 1u);   // round-to-nearest-even
  return (unsigned short)(r >> 16);
}

static constexpr int Bn = 8, Cc = 64, Hh = 128, Ww = 128, KK = 9;
static constexpr int NPIX = Bn * Hh * Ww;                 // 131072

// ---------------------------------------------------------------------------
// Pack all four weight tensors OIHW f32 -> [p][o][c] bf16 (A-GEMM friendly).
// ---------------------------------------------------------------------------
__global__ __launch_bounds__(256) void pack_w_kernel(
    const float* __restrict__ w0, const float* __restrict__ w1,
    const float* __restrict__ w2, const float* __restrict__ w3,
    unsigned short* __restrict__ o0, unsigned short* __restrict__ o1,
    unsigned short* __restrict__ o2, unsigned short* __restrict__ o3) {
  int idx = blockIdx.x * 256 + threadIdx.x;
  if (idx >= Cc * Cc * KK) return;
  int p = idx % KK, c = (idx / KK) % Cc, o = idx / (KK * Cc);
  int src = (o * Cc + c) * KK + p;
  int dst = (p * Cc + o) * Cc + c;
  o0[dst] = f2bf(w0[src]); o1[dst] = f2bf(w1[src]);
  o2[dst] = f2bf(w2[src]); o3[dst] = f2bf(w3[src]);
}

// ---------------------------------------------------------------------------
// Stage-1 prep: x,edge (NCHW f32) -> NHWC bf16 copies, plus Gaussian PAC
// kernel kern1[b][p][h][w] = exp(-0.5 * sum_c (g_patch - g_center)^2)
// (zero-padded patches, matching jnp.pad semantics).
// ---------------------------------------------------------------------------
__global__ __launch_bounds__(256) void prep1_kernel(
    const float* __restrict__ x, const float* __restrict__ g,
    unsigned short* __restrict__ xbf, unsigned short* __restrict__ gbf,
    float* __restrict__ kern) {
  int t = blockIdx.x * 256 + threadIdx.x;          // pixel id = (b*H+h)*W+w
  int w = t & (Ww - 1), h = (t >> 7) & (Hh - 1), b = t >> 14;
  float gc[Cc];
#pragma unroll
  for (int c = 0; c < Cc; ++c) {
    size_t a = ((size_t)(b * Cc + c) << 14) + (h << 7) + w;
    float xv = x[a], gv = g[a];
    xbf[(size_t)t * 64 + c] = f2bf(xv);
    gbf[(size_t)t * 64 + c] = f2bf(gv);
    gc[c] = gv;
  }
  for (int p = 0; p < KK; ++p) {
    int dy = p / 3 - 1, dx = p % 3 - 1;
    int hh = h + dy, ww = w + dx;
    bool inb = (hh >= 0) && (hh < Hh) && (ww >= 0) && (ww < Ww);
    float d2 = 0.f;
    if (inb) {
#pragma unroll
      for (int c = 0; c < Cc; ++c) {
        float gv = g[((size_t)(b * Cc + c) << 14) + (hh << 7) + ww];
        float d = gv - gc[c]; d2 += d * d;
      }
    } else {
#pragma unroll
      for (int c = 0; c < Cc; ++c) d2 += gc[c] * gc[c];
    }
    kern[((size_t)(b * KK + p) << 14) + (h << 7) + w] = __expf(-0.5f * d2);
  }
}

// ---------------------------------------------------------------------------
// Stage-2 PAC kernel from the relu'd edge feature (NHWC bf16 guide).
// ---------------------------------------------------------------------------
__global__ __launch_bounds__(256) void kern2_kernel(
    const unsigned short* __restrict__ g, float* __restrict__ kern) {
  int t = blockIdx.x * 256 + threadIdx.x;
  int w = t & (Ww - 1), h = (t >> 7) & (Hh - 1), b = t >> 14;
  float gc[Cc];
#pragma unroll
  for (int c = 0; c < Cc; ++c) gc[c] = bf2f(g[(size_t)t * 64 + c]);
  for (int p = 0; p < KK; ++p) {
    int dy = p / 3 - 1, dx = p % 3 - 1;
    int hh = h + dy, ww = w + dx;
    bool inb = (hh >= 0) && (hh < Hh) && (ww >= 0) && (ww < Ww);
    float d2 = 0.f;
    if (inb) {
      size_t nb = (((size_t)((b << 7) + hh) << 7) + ww) * 64;
#pragma unroll
      for (int c = 0; c < Cc; ++c) {
        float d = bf2f(g[nb + c]) - gc[c]; d2 += d * d;
      }
    } else {
#pragma unroll
      for (int c = 0; c < Cc; ++c) d2 += gc[c] * gc[c];
    }
    kern[((size_t)(b * KK + p) << 14) + (h << 7) + w] = __expf(-0.5f * d2);
  }
}

// ---------------------------------------------------------------------------
// Implicit-GEMM conv, one workgroup per (b,h) row: 8 waves x 16 pixels,
// all 64 output channels per wave (4 m-tiles of v8f accum).
// K = 9 taps x 64 Cin, chunked 32 -> 72 v_wmma_f32_16x16x32_bf16 per wave.
// PAC: per-tap f32 accumulator folded with acc += kern * tacc.
// FINAL: residual-add -> f32 NCHW (d_out); else relu -> bf16 NHWC.
// ---------------------------------------------------------------------------
template <bool PAC, bool FINAL>
__global__ __launch_bounds__(256) void pac_gemm_kernel(
    const unsigned short* __restrict__ in,   // NHWC bf16 activations
    const float* __restrict__ kern,          // [B][9][H][W] f32 (PAC only)
    const unsigned short* __restrict__ wA,   // [9][64][64] bf16 weights
    const float* __restrict__ bias,          // [64] f32
    const float* __restrict__ resid,         // NCHW f32 (FINAL only)
    unsigned short* __restrict__ outBF,      // NHWC bf16 (!FINAL)
    float* __restrict__ outF)                // NCHW f32 (FINAL)
{
  __shared__ unsigned short wLds[2][Cc * Cc];  // double-buffered 8KB tap slices
  const int tid  = threadIdx.x;
  const int lane = tid & 31, wave = tid >> 5;
  const int half = lane >> 4;                // lanes 0-15 vs 16-31
  const int m    = lane & 15;                // A row (M) and B column (N)
  const int b    = blockIdx.x >> 7;
  const int h    = blockIdx.x & 127;
  const int wpix = wave * 16 + m;            // this lane's output pixel

  // ---- tap-slice stager: async global->LDS when available ----
  auto stage = [&](int p, int buf) {
    const uint4* s = (const uint4*)(wA + p * Cc * Cc);
    uint4* d = (uint4*)wLds[buf];
#if USE_ASYNC_LDS
    // builtin signature: (v4i addrspace(1)*, v4i addrspace(3)*, imm off, imm cpol)
    __builtin_amdgcn_global_load_async_to_lds_b128(
        (__attribute__((address_space(1))) v4i*)(unsigned long long)(s + tid),
        (__attribute__((address_space(3))) v4i*)(unsigned int)(unsigned long long)(d + tid),
        0, 0);
    __builtin_amdgcn_global_load_async_to_lds_b128(
        (__attribute__((address_space(1))) v4i*)(unsigned long long)(s + tid + 256),
        (__attribute__((address_space(3))) v4i*)(unsigned int)(unsigned long long)(d + tid + 256),
        0, 0);
#else
    d[tid] = s[tid];
    d[tid + 256] = s[tid + 256];
#endif
  };

  v8f acc[4] = {};

  stage(0, 0);                               // prologue: slice for tap 0
  for (int p = 0; p < KK; ++p) {
#if USE_ASYNC_LDS
#if __has_builtin(__builtin_amdgcn_s_wait_asynccnt)
    __builtin_amdgcn_s_wait_asynccnt(0);
#else
    asm volatile("s_wait_asynccnt 0x0" ::: "memory");
#endif
#endif
    __syncthreads();                         // slice p landed + prev compute done
    if (p + 1 < KK) stage(p + 1, (p + 1) & 1);   // overlaps with WMMAs below
    const unsigned short* wl = wLds[p & 1];

    const int dy = p / 3 - 1, dx = p % 3 - 1;
    const int hh = h + dy, ww = wpix + dx;
    const bool inb = (hh >= 0) && (hh < Hh) && (ww >= 0) && (ww < Ww);
    float kv = 0.f;
    if (PAC) kv = kern[((size_t)(b * KK + p) << 14) + (h << 7) + wpix];

    v8f tacc[4];
    if (PAC) {
#pragma unroll
      for (int mt = 0; mt < 4; ++mt) tacc[mt] = (v8f)0.f;
    }
    v8f* dst = PAC ? tacc : acc;

#pragma unroll
    for (int chunk = 0; chunk < 2; ++chunk) {
      const int c0 = chunk * 32;
      // ---- B fragment: 16 consecutive channels (K ascending) per lane ----
      BF16x16 bfr;
      if (inb) {
        const uint4* q = (const uint4*)(
            in + ((((size_t)((b << 7) + hh) << 7) + ww) * 64 + c0 + half * 16));
        bfr.q[0] = q[0]; bfr.q[1] = q[1];
      } else {
        bfr.q[0] = make_uint4(0, 0, 0, 0);
        bfr.q[1] = make_uint4(0, 0, 0, 0);
      }
      // ---- hoist all 4 A fragments, then 4 back-to-back WMMAs ----
      BF16x16 af[4];
#pragma unroll
      for (int mt = 0; mt < 4; ++mt) {
        const int ao = (mt * 16 + m) * Cc + c0 + half * 16;  // per-lane K halves
        af[mt].q[0] = *(const uint4*)(wl + ao);
        af[mt].q[1] = *(const uint4*)(wl + ao + 8);
      }
#pragma unroll
      for (int mt = 0; mt < 4; ++mt) {
        dst[mt] = __builtin_amdgcn_wmma_f32_16x16x32_bf16(
            false, af[mt].v, false, bfr.v, (short)0, dst[mt], false, false);
      }
    }
    if (PAC) {                               // fold tap with f32 kern weight
#pragma unroll
      for (int mt = 0; mt < 4; ++mt) acc[mt] += tacc[mt] * kv;
    }
  }

  // ---- epilogue: C/D layout VGPR j <-> M = j + 8*half, N = lane&15 ----
#pragma unroll
  for (int mt = 0; mt < 4; ++mt) {
    if (FINAL) {
#pragma unroll
      for (int j = 0; j < 8; ++j) {
        int o = mt * 16 + half * 8 + j;
        float v = acc[mt][j] + bias[o];
        size_t a = ((((size_t)(b * Cc + o)) << 7) + h);
        a = (a << 7) + wpix;                 // NCHW flat index
        outF[a] = v + resid[a];
      }
    } else {
      unsigned short pk[8];
#pragma unroll
      for (int j = 0; j < 8; ++j) {
        int o = mt * 16 + half * 8 + j;
        float v = acc[mt][j] + bias[o];
        pk[j] = f2bf(v > 0.f ? v : 0.f);     // relu
      }
      size_t base = (((size_t)blockIdx.x) * Ww + wpix) * 64 + mt * 16 + half * 8;
      *(uint4*)(outBF + base) = *(const uint4*)pk;   // 8 channels, one b128
    }
  }
}

// ---------------------------------------------------------------------------
extern "C" void kernel_launch(void* const* d_in, const int* in_sizes, int n_in,
                              void* d_out, int out_size, void* d_ws, size_t ws_size,
                              hipStream_t stream) {
  (void)in_sizes; (void)n_in; (void)out_size; (void)ws_size;
  const float* x      = (const float*)d_in[0];
  const float* edge   = (const float*)d_in[1];
  const float* w_pac1 = (const float*)d_in[2];
  const float* b_pac1 = (const float*)d_in[3];
  const float* w_pac2 = (const float*)d_in[4];
  const float* b_pac2 = (const float*)d_in[5];
  const float* w_e1   = (const float*)d_in[6];
  const float* b_e1   = (const float*)d_in[7];
  const float* w_e2   = (const float*)d_in[8];
  const float* b_e2   = (const float*)d_in[9];

  float* out_sr   = (float*)d_out;
  float* out_edge = out_sr + (size_t)Bn * Cc * Hh * Ww;

  char* ws = (char*)d_ws;
  size_t off = 0;
  auto take = [&](size_t bytes) {
    char* p = ws + off; off = (off + bytes + 255) & ~(size_t)255; return p;
  };
  unsigned short* xbf   = (unsigned short*)take((size_t)NPIX * 64 * 2);
  unsigned short* ebf   = (unsigned short*)take((size_t)NPIX * 64 * 2);
  unsigned short* sr1   = (unsigned short*)take((size_t)NPIX * 64 * 2);
  unsigned short* e1    = (unsigned short*)take((size_t)NPIX * 64 * 2);
  float*          kern1 = (float*)take((size_t)Bn * KK * Hh * Ww * 4);
  float*          kern2 = (float*)take((size_t)Bn * KK * Hh * Ww * 4);
  unsigned short* wA1   = (unsigned short*)take((size_t)KK * Cc * Cc * 2);
  unsigned short* wAe1  = (unsigned short*)take((size_t)KK * Cc * Cc * 2);
  unsigned short* wA2   = (unsigned short*)take((size_t)KK * Cc * Cc * 2);
  unsigned short* wAe2  = (unsigned short*)take((size_t)KK * Cc * Cc * 2);

  // 1) pack all weights to bf16 [p][o][c]
  pack_w_kernel<<<(Cc * Cc * KK + 255) / 256, 256, 0, stream>>>(
      w_pac1, w_e1, w_pac2, w_e2, wA1, wAe1, wA2, wAe2);
  // 2) bf16 NHWC copies + PAC kernel 1 (guide = edge)
  prep1_kernel<<<NPIX / 256, 256, 0, stream>>>(x, edge, xbf, ebf, kern1);
  // 3) res_sr1 = relu(pac_conv(x | kern1, w_pac1) + b)
  pac_gemm_kernel<true, false><<<Bn * Hh, 256, 0, stream>>>(
      xbf, kern1, wA1, b_pac1, nullptr, sr1, nullptr);
  // 4) res_edge1 = relu(conv(edge, w_e1) + b)
  pac_gemm_kernel<false, false><<<Bn * Hh, 256, 0, stream>>>(
      ebf, nullptr, wAe1, b_e1, nullptr, e1, nullptr);
  // 5) PAC kernel 2 (guide = res_edge1)
  kern2_kernel<<<NPIX / 256, 256, 0, stream>>>(e1, kern2);
  // 6) out_sr = x + pac_conv(res_sr1 | kern2, w_pac2) + b
  pac_gemm_kernel<true, true><<<Bn * Hh, 256, 0, stream>>>(
      sr1, kern2, wA2, b_pac2, x, nullptr, out_sr);
  // 7) out_edge = edge + conv(res_edge1, w_e2) + b
  pac_gemm_kernel<false, true><<<Bn * Hh, 256, 0, stream>>>(
      e1, nullptr, wAe2, b_e2, edge, nullptr, out_edge);
}
